// EncoderLayer_37082747633774
// MI455X (gfx1250) — matrix operand order repaired and checked by reading, not compiled, not run on hardware
//
#include <hip/hip_runtime.h>
#include <hip/hip_bf16.h>

typedef __bf16 bf16;
typedef __bf16 v16bf __attribute__((ext_vector_type(16)));
typedef float  v8f   __attribute__((ext_vector_type(8)));

#define D_MODEL 1024
#define NHEAD   16
#define D_K     64
#define D_FF    4096
#define BATCH   4
#define SEQ     2048
#define MROWS   (BATCH * SEQ)   // 8192

// ---------------------------------------------------------------------------
// DPP16 row_xmask butterfly within 16-lane rows (VALU, no LDS round-trip).
// ---------------------------------------------------------------------------
template <int MASK>
__device__ __forceinline__ float xor16_dpp(float v) {
  int s = __builtin_bit_cast(int, v);
  int r = __builtin_amdgcn_update_dpp(s, s, 0x160 + MASK, 0xF, 0xF, false);
  return __builtin_bit_cast(float, r);
}
__device__ __forceinline__ float rowmax16(float v) {
  v = fmaxf(v, xor16_dpp<1>(v));
  v = fmaxf(v, xor16_dpp<2>(v));
  v = fmaxf(v, xor16_dpp<4>(v));
  v = fmaxf(v, xor16_dpp<8>(v));
  return v;
}
__device__ __forceinline__ float rowsum16(float v) {
  v += xor16_dpp<1>(v);
  v += xor16_dpp<2>(v);
  v += xor16_dpp<4>(v);
  v += xor16_dpp<8>(v);
  return v;
}

// ---------------------------------------------------------------------------
// Cast fp32 weight [K][N] -> bf16 transposed [N][K]
// ---------------------------------------------------------------------------
__global__ void cast_transpose_kernel(const float* __restrict__ in,
                                      bf16* __restrict__ out, int K, int N) {
  int i = blockIdx.x * 256 + threadIdx.x;   // K*N <= 4M, fits int
  int k = i / N;
  int n = i - k * N;
  out[(size_t)n * K + k] = (bf16)in[i];
}

// ---------------------------------------------------------------------------
// LayerNorm (scalar alpha/beta, unbiased std, eps added to std) -> bf16
// ---------------------------------------------------------------------------
__global__ void layernorm_kernel(const float* __restrict__ x,
                                 const float* __restrict__ alpha,
                                 const float* __restrict__ beta,
                                 bf16* __restrict__ out) {
  __shared__ float sh1[256];
  __shared__ float sh2[256];
  int row = blockIdx.x;
  int tid = threadIdx.x;
  const float* xr = x + (size_t)row * D_MODEL;
  float v[4];
  float s1 = 0.f, s2 = 0.f;
  for (int i = 0; i < 4; ++i) {
    v[i] = xr[tid + i * 256];
    s1 += v[i];
    s2 += v[i] * v[i];
  }
  sh1[tid] = s1; sh2[tid] = s2;
  __syncthreads();
  for (int off = 128; off > 0; off >>= 1) {
    if (tid < off) { sh1[tid] += sh1[tid + off]; sh2[tid] += sh2[tid + off]; }
    __syncthreads();
  }
  float mean = sh1[0] * (1.0f / 1024.0f);
  float var  = (sh2[0] - 1024.0f * mean * mean) * (1.0f / 1023.0f);
  float inv  = alpha[0] / (sqrtf(fmaxf(var, 0.f)) + 1e-6f);
  float bet  = beta[0];
  for (int i = 0; i < 4; ++i)
    out[(size_t)row * D_MODEL + tid + i * 256] = (bf16)((v[i] - mean) * inv + bet);
}

// ---------------------------------------------------------------------------
// Repack V [B,S,H*Dk] (bf16) -> Vt [B,H,Dk,S] (bf16)
// ---------------------------------------------------------------------------
__global__ void v_transpose_kernel(const bf16* __restrict__ v,
                                   bf16* __restrict__ vt) {
  int i = blockIdx.x * 256 + threadIdx.x;      // B*H*Dk*S = 8388608
  int s = i & (SEQ - 1);
  int r = i >> 11;
  int d = r & (D_K - 1); r >>= 6;
  int h = r & (NHEAD - 1);
  int b = r >> 4;
  vt[i] = v[(size_t)(b * SEQ + s) * D_MODEL + h * D_K + d];
}

// ---------------------------------------------------------------------------
// Tiled bf16 GEMM, fp32 accumulation via v_wmma_f32_16x16x32_bf16.
// Double-buffered LDS staging with GLOBAL_LOAD_ASYNC_TO_LDS_B128:
// prefetch tile t+1 (ASYNCcnt) while WMMAs consume tile t; one barrier/iter.
// 256 threads = 8 waves; block tile 128x128; wave tile 32x64 (2x4 WMMA).
// ---------------------------------------------------------------------------
template <bool RELU, bool OUT_BF16, bool RES>
__global__ void gemm_bf16_kernel(const bf16* __restrict__ A,
                                 const bf16* __restrict__ Bt,
                                 const float* __restrict__ bias,
                                 const float* __restrict__ resid,
                                 void* __restrict__ out,
                                 int M, int N, int K) {
  __shared__ alignas(16) bf16 As[2][128][40];
  __shared__ alignas(16) bf16 Bs[2][128][40];

  int tid  = threadIdx.x;
  int w    = tid >> 5;
  int lane = tid & 31;
  int wm   = w & 3;          // 4 waves along M
  int wn   = w >> 2;         // 2 waves along N
  int m0   = blockIdx.y * 128;
  int n0   = blockIdx.x * 128;
  int lcol = lane & 15;
  int hi   = (lane >= 16);

  auto issue_tile = [&](int kt, int buf) {
    for (int i = 0; i < 2; ++i) {
      int c   = tid + i * 256;
      int row = c >> 2;
      int cc  = (c & 3) * 8;
      const bf16* ga = A  + (size_t)(m0 + row) * K + kt + cc;
      const bf16* gb = Bt + (size_t)(n0 + row) * K + kt + cc;
      unsigned la = (unsigned)(size_t)&As[buf][row][cc];   // LDS byte offset
      unsigned lb = (unsigned)(size_t)&Bs[buf][row][cc];
      asm volatile("global_load_async_to_lds_b128 %0, %1, off"
                   :: "v"(la), "v"(ga) : "memory");
      asm volatile("global_load_async_to_lds_b128 %0, %1, off"
                   :: "v"(lb), "v"(gb) : "memory");
    }
  };

  v8f acc[2][4] = {};

  const int nt = K / 32;
  issue_tile(0, 0);
  asm volatile("s_wait_asynccnt 0" ::: "memory");
  __syncthreads();

  for (int t = 0; t < nt; ++t) {
    int buf = t & 1;
    if (t + 1 < nt) issue_tile((t + 1) * 32, (t + 1) & 1);  // overlap with compute

    v16bf af[2], bfm[4];
    {
      int ab = hi ? 8 : 0;       // A-frag: lane<16 -> K{0..7,16..23}; else {8..15,24..31}
      for (int mi = 0; mi < 2; ++mi) {
        const bf16* p = &As[buf][wm * 32 + mi * 16 + lcol][ab];
        ((uint4*)&af[mi])[0] = *(const uint4*)p;
        ((uint4*)&af[mi])[1] = *(const uint4*)(p + 16);
      }
      int bb = hi ? 8 : 0;       // B-frag: lane<16 -> K{0..7,16..23}; else {8..15,24..31}
      for (int ni = 0; ni < 4; ++ni) {
        const bf16* p = &Bs[buf][wn * 64 + ni * 16 + lcol][bb];
        ((uint4*)&bfm[ni])[0] = *(const uint4*)p;
        ((uint4*)&bfm[ni])[1] = *(const uint4*)(p + 16);
      }
    }
    for (int ni = 0; ni < 4; ++ni)
      for (int mi = 0; mi < 2; ++mi)
        acc[mi][ni] = __builtin_amdgcn_wmma_f32_16x16x32_bf16(
            false, af[mi], false, bfm[ni], (short)0, acc[mi][ni], false, false);

    asm volatile("s_wait_asynccnt 0" ::: "memory");  // prefetch landed
    __syncthreads();                                 // all waves done reading buf
  }

  // epilogue: C layout -> VGPR j holds rows j (lanes 0-15) and j+8 (lanes 16-31)
  for (int mi = 0; mi < 2; ++mi) {
    int mbase = m0 + wm * 32 + mi * 16 + (hi ? 8 : 0);
    for (int ni = 0; ni < 4; ++ni) {
      int n  = n0 + wn * 64 + ni * 16 + lcol;
      float bv = bias[n];
      for (int j = 0; j < 8; ++j) {
        int m = mbase + j;
        float val = acc[mi][ni][j] + bv;
        if (RELU) val = fmaxf(val, 0.f);
        if (RES)  val += resid[(size_t)m * N + n];
        if (OUT_BF16) ((bf16*)out)[(size_t)m * N + n] = (bf16)val;
        else          ((float*)out)[(size_t)m * N + n] = val;
      }
    }
  }
}

// ---------------------------------------------------------------------------
// Flash attention: one wave per 16-query tile of one (b,h).
// Per 32-key chunk: 4 score WMMAs (2 halves x K=64 as 2xK32), mask+scale, online softmax
// with DPP row_xmask reductions, P staged via LDS to A-fragment layout,
// then 4 PV WMMAs against Vt.
// ---------------------------------------------------------------------------
__global__ void attention_kernel(const bf16* __restrict__ q,
                                 const bf16* __restrict__ k,
                                 const bf16* __restrict__ vt,
                                 const int* __restrict__ mask,
                                 bf16* __restrict__ ctx) {
  __shared__ alignas(16) bf16 P[8][16][40];

  int tid  = threadIdx.x;
  int w    = tid >> 5;
  int lane = tid & 31;
  int gid  = blockIdx.x * 8 + w;        // 0 .. 8191
  int qt   = gid & 127;                 // S/16 tiles
  int bh   = gid >> 7;
  int h    = bh & (NHEAD - 1);
  int b    = bh >> 4;
  int q0   = qt * 16;
  int lcol = lane & 15;
  int hi   = (lane >= 16);

  // Q fragments: d 0..31 and d 32..63
  v16bf qf[2];
  {
    const bf16* base = q + ((size_t)(b * SEQ + q0 + lcol) * D_MODEL + h * D_K);
    const bf16* p0 = base + (hi ? 8 : 0);
    ((uint4*)&qf[0])[0] = *(const uint4*)p0;
    ((uint4*)&qf[0])[1] = *(const uint4*)(p0 + 16);
    const bf16* p1 = base + 32 + (hi ? 8 : 0);
    ((uint4*)&qf[1])[0] = *(const uint4*)p1;
    ((uint4*)&qf[1])[1] = *(const uint4*)(p1 + 16);
  }

  v8f o[4] = {};
  float rm[8], rl[8];
  for (int j = 0; j < 8; ++j) { rm[j] = -3.0e38f; rl[j] = 0.f; }
  const float scale = 0.125f;           // 1/sqrt(64)

  for (int kb = 0; kb < SEQ; kb += 32) {
    // scores for two 16-key halves
    v8f s[2];
    for (int half = 0; half < 2; ++half) {
      int key = kb + half * 16 + lcol;
      const bf16* kbase = k + ((size_t)(b * SEQ + key) * D_MODEL + h * D_K + (hi ? 8 : 0));
      v16bf kf0, kf1;
      ((uint4*)&kf0)[0] = *(const uint4*)kbase;
      ((uint4*)&kf0)[1] = *(const uint4*)(kbase + 16);
      ((uint4*)&kf1)[0] = *(const uint4*)(kbase + 32);
      ((uint4*)&kf1)[1] = *(const uint4*)(kbase + 48);
      v8f z = {};
      z = __builtin_amdgcn_wmma_f32_16x16x32_bf16(false, qf[0], false, kf0, (short)0, z, false, false);
      z = __builtin_amdgcn_wmma_f32_16x16x32_bf16(false, qf[1], false, kf1, (short)0, z, false, false);
      s[half] = z;
    }

    int msk0 = mask[b * SEQ + kb + lcol];
    int msk1 = mask[b * SEQ + kb + 16 + lcol];

    for (int j = 0; j < 8; ++j) {
      float s0 = s[0][j] * scale; if (msk0 == 0) s0 = -1e9f;
      float s1 = s[1][j] * scale; if (msk1 == 0) s1 = -1e9f;
      float cm = rowmax16(fmaxf(s0, s1));
      float nm = fmaxf(rm[j], cm);
      float f  = __expf(rm[j] - nm);
      float p0 = __expf(s0 - nm);
      float p1 = __expf(s1 - nm);
      float ps = rowsum16(p0 + p1);
      rl[j] = rl[j] * f + ps;
      rm[j] = nm;
      for (int ni = 0; ni < 4; ++ni) o[ni][j] *= f;
      int row = j + (hi ? 8 : 0);
      P[w][row][lcol]      = (bf16)p0;
      P[w][row][lcol + 16] = (bf16)p1;
    }
    asm volatile("s_wait_dscnt 0" ::: "memory");

    // P -> A fragment (16 queries x 32 keys)
    v16bf pf;
    {
      const bf16* pp = &P[w][lcol][hi ? 8 : 0];
      ((uint4*)&pf)[0] = *(const uint4*)pp;
      ((uint4*)&pf)[1] = *(const uint4*)(pp + 16);
    }
    // Vt fragments and PV accumulation
    for (int ni = 0; ni < 4; ++ni) {
      const bf16* vb = vt + ((size_t)((b * NHEAD + h) * D_K + ni * 16 + lcol)) * SEQ
                          + kb + (hi ? 8 : 0);
      v16bf vf;
      ((uint4*)&vf)[0] = *(const uint4*)vb;
      ((uint4*)&vf)[1] = *(const uint4*)(vb + 16);
      o[ni] = __builtin_amdgcn_wmma_f32_16x16x32_bf16(
          false, pf, false, vf, (short)0, o[ni], false, false);
    }
  }

  // normalize + write ctx (bf16, [B,S,H*Dk])
  for (int ni = 0; ni < 4; ++ni) {
    for (int j = 0; j < 8; ++j) {
      int row = q0 + j + (hi ? 8 : 0);
      int d   = h * D_K + ni * 16 + lcol;
      ctx[(size_t)(b * SEQ + row) * D_MODEL + d] = (bf16)(o[ni][j] / rl[j]);
    }
  }
}

// ---------------------------------------------------------------------------
extern "C" void kernel_launch(void* const* d_in, const int* in_sizes, int n_in,
                              void* d_out, int out_size, void* d_ws, size_t ws_size,
                              hipStream_t stream) {
  (void)in_sizes; (void)n_in; (void)out_size; (void)ws_size;

  const float* x    = (const float*)d_in[0];
  const int*   msk  = (const int*)d_in[1];
  const float* wq   = (const float*)d_in[2];
  const float* bq   = (const float*)d_in[3];
  const float* wk   = (const float*)d_in[4];
  const float* bk   = (const float*)d_in[5];
  const float* wv   = (const float*)d_in[6];
  const float* bv   = (const float*)d_in[7];
  const float* wo   = (const float*)d_in[8];
  const float* bo   = (const float*)d_in[9];
  const float* w1   = (const float*)d_in[10];
  const float* b1   = (const float*)d_in[11];
  const float* w2   = (const float*)d_in[12];
  const float* b2   = (const float*)d_in[13];
  const float* a1   = (const float*)d_in[14];
  const float* be1  = (const float*)d_in[15];
  const float* a2   = (const float*)d_in[16];
  const float* be2  = (const float*)d_in[17];
  float* out = (float*)d_out;

  char* ws = (char*)d_ws;
  const size_t MB = (size_t)1 << 20;
  bf16*  xn  = (bf16*)(ws + 0 * MB);      // 16MB (reused as ctx later)
  bf16*  qb  = (bf16*)(ws + 16 * MB);     // 16MB
  bf16*  kb  = (bf16*)(ws + 32 * MB);     // 16MB
  bf16*  vb  = (bf16*)(ws + 48 * MB);     // 16MB
  bf16*  vtb = (bf16*)(ws + 64 * MB);     // 16MB
  float* x2  = (float*)(ws + 80 * MB);    // 32MB (fp32)
  bf16*  xn2 = (bf16*)(ws + 112 * MB);    // 16MB
  bf16*  hb  = (bf16*)(ws + 128 * MB);    // 64MB
  bf16*  wqt = (bf16*)(ws + 192 * MB);    // 2MB each
  bf16*  wkt = (bf16*)(ws + 194 * MB);
  bf16*  wvt = (bf16*)(ws + 196 * MB);
  bf16*  wot = (bf16*)(ws + 198 * MB);
  bf16*  w1t = (bf16*)(ws + 200 * MB);    // 8MB
  bf16*  w2t = (bf16*)(ws + 208 * MB);    // 8MB
  bf16*  ctx = xn;                        // reuse (xn dead after V projection)

  // weights -> bf16 transposed
  cast_transpose_kernel<<<(D_MODEL * D_MODEL) / 256, 256, 0, stream>>>(wq, wqt, D_MODEL, D_MODEL);
  cast_transpose_kernel<<<(D_MODEL * D_MODEL) / 256, 256, 0, stream>>>(wk, wkt, D_MODEL, D_MODEL);
  cast_transpose_kernel<<<(D_MODEL * D_MODEL) / 256, 256, 0, stream>>>(wv, wvt, D_MODEL, D_MODEL);
  cast_transpose_kernel<<<(D_MODEL * D_MODEL) / 256, 256, 0, stream>>>(wo, wot, D_MODEL, D_MODEL);
  cast_transpose_kernel<<<(D_MODEL * D_FF) / 256, 256, 0, stream>>>(w1, w1t, D_MODEL, D_FF);
  cast_transpose_kernel<<<(D_FF * D_MODEL) / 256, 256, 0, stream>>>(w2, w2t, D_FF, D_MODEL);

  // LN1
  layernorm_kernel<<<MROWS, 256, 0, stream>>>(x, a1, be1, xn);

  // Q, K, V projections (bf16 out)
  dim3 gqkv(D_MODEL / 128, MROWS / 128);
  gemm_bf16_kernel<false, true, false><<<gqkv, 256, 0, stream>>>(xn, wqt, bq, nullptr, qb, MROWS, D_MODEL, D_MODEL);
  gemm_bf16_kernel<false, true, false><<<gqkv, 256, 0, stream>>>(xn, wkt, bk, nullptr, kb, MROWS, D_MODEL, D_MODEL);
  gemm_bf16_kernel<false, true, false><<<gqkv, 256, 0, stream>>>(xn, wvt, bv, nullptr, vb, MROWS, D_MODEL, D_MODEL);

  // V -> Vt [B,H,Dk,S]
  v_transpose_kernel<<<(BATCH * NHEAD * D_K * SEQ) / 256, 256, 0, stream>>>(vb, vtb);

  // attention: 8192 wave-tiles, 8 per block
  attention_kernel<<<1024, 256, 0, stream>>>(qb, kb, vtb, msk, ctx);

  // O projection + residual -> x2 (fp32)
  gemm_bf16_kernel<false, false, true><<<gqkv, 256, 0, stream>>>(ctx, wot, bo, x, x2, MROWS, D_MODEL, D_MODEL);

  // LN2
  layernorm_kernel<<<MROWS, 256, 0, stream>>>(x2, a2, be2, xn2);

  // FFN1 (ReLU, bf16 out)
  dim3 gff1(D_FF / 128, MROWS / 128);
  gemm_bf16_kernel<true, true, false><<<gff1, 256, 0, stream>>>(xn2, w1t, b1, nullptr, hb, MROWS, D_FF, D_MODEL);

  // FFN2 + residual -> out (fp32)
  gemm_bf16_kernel<false, false, true><<<gqkv, 256, 0, stream>>>(hb, w2t, b2, x2, out, MROWS, D_MODEL, D_FF);
}
